// GIN_36120674959489
// MI455X (gfx1250) — matrix operand order repaired and checked by reading, not compiled, not run on hardware
//
#include <hip/hip_runtime.h>
#include <hip/hip_bf16.h>
#include <stdint.h>

#define N_NODES 50000
#define N_EDGES 800000
#define NFEAT   256
#define NHID    512
#define BN_EPS  1e-5f
#define LDS_PAD 40   // row stride (bf16 elems): 80B rows, 16B aligned, de-conflicted b128 reads

typedef __attribute__((ext_vector_type(16))) __bf16 v16bf;
typedef __attribute__((ext_vector_type(8)))  float  v8f;

// ---------------------------------------------------------------- h0 = x
__global__ void copy_x_kernel(const float* __restrict__ x, float* __restrict__ h0) {
    int i = blockIdx.x * blockDim.x + threadIdx.x;
    if (i < (N_NODES * NFEAT / 4)) ((float4*)h0)[i] = ((const float4*)x)[i];
}

// --------------------------------------------- h0[dst] += x[src] per edge
__global__ void edge_scatter_kernel(const float* __restrict__ x,
                                    const long long* __restrict__ ei,
                                    float* __restrict__ h0) {
    const int tid = threadIdx.x;
    const long long e = (long long)blockIdx.x * 4 + (tid >> 6);   // 4 edges / block
    if (e >= N_EDGES) return;
    const int src = (int)ei[e];
    const int dst = (int)ei[N_EDGES + e];
    const int f = (tid & 63) * 4;
    const float4 v = *(const float4*)(x + (size_t)src * NFEAT + f);
    float* d = h0 + (size_t)dst * NFEAT + f;
    atomicAdd(d + 0, v.x); atomicAdd(d + 1, v.y);
    atomicAdd(d + 2, v.z); atomicAdd(d + 3, v.w);
}

// ---------------------------------------------------------------- WMMA GEMM
// C[M x N] = epilogue( A[M x K] @ W[K x N] + bias ).
// Block tile 128(M) x 64(N) x 32(K); 8 wave32 waves in a 4x2 grid;
// each wave owns a 32x32 register tile = 2x2 WMMA subtiles (4 WMMAs / K-step).
// LDS tiles are double-buffered: one barrier per K-step, global loads for
// tile t+1 overlap the WMMAs on tile t. fp32->bf16 staging uses the native
// hardware converter (plain __bf16 casts -> v_cvt_pk_bf16_f32 class ops).
// BN_IN: fold a = a*scale[k]+shift[k] into the A staging.
template<int K, bool BN_IN, bool RELU>
__global__ __launch_bounds__(256)
void gemm_wmma_kernel(const float* __restrict__ A,
                      const float* __restrict__ W,
                      const float* __restrict__ bias,
                      const float* __restrict__ scale,
                      const float* __restrict__ shift,
                      float* __restrict__ C,
                      int M, int N)
{
    __shared__ __align__(16) __bf16 sA [2][128][LDS_PAD];  // bf16 A tile  [m][k]
    __shared__ __align__(16) __bf16 sBt[2][ 64][LDS_PAD];  // bf16 B tile, transposed [n][k]

    const int tid  = threadIdx.x;
    const int wave = tid >> 5;
    const int lane = tid & 31;
    const int g    = lane >> 4;        // ISA half-wave group (lanes 0-15 / 16-31)
    const int l16  = lane & 15;

    const int mBlock = blockIdx.x * 128;
    const int nBlock = blockIdx.y * 64;

    const int mw = wave >> 1;          // 0..3 : 32-row band
    const int nw = wave & 1;           // 0..1 : 32-col band

    v8f acc[2][2] = {{{}, {}}, {{}, {}}};

    // global->reg staging indices
    const int arow = tid >> 1;              // 0..127
    const int acol = (tid & 1) * 16;        // 0 or 16
    const int bk   = tid >> 3;              // 0..31
    const int bn   = (tid & 7) * 8;         // 0..56

    float av[16];                           // A staging regs (one 128x32 tile / block)
    float bv[8];                            // B staging regs

    auto load_tile = [&](int k0) {
        const int row = mBlock + arow;
        if (row < M) {
            const float* p = A + (size_t)row * K + k0 + acol;
            const float4 u0 = *(const float4*)(p);
            const float4 u1 = *(const float4*)(p + 4);
            const float4 u2 = *(const float4*)(p + 8);
            const float4 u3 = *(const float4*)(p + 12);
            av[0]=u0.x;  av[1]=u0.y;  av[2]=u0.z;  av[3]=u0.w;
            av[4]=u1.x;  av[5]=u1.y;  av[6]=u1.z;  av[7]=u1.w;
            av[8]=u2.x;  av[9]=u2.y;  av[10]=u2.z; av[11]=u2.w;
            av[12]=u3.x; av[13]=u3.y; av[14]=u3.z; av[15]=u3.w;
            if (k0 + 32 < K) __builtin_prefetch(p + 32, 0, 0);
        } else {
            #pragma unroll
            for (int i = 0; i < 16; ++i) av[i] = 0.f;
        }
        if constexpr (BN_IN) {
            #pragma unroll
            for (int i = 0; i < 16; ++i)
                av[i] = av[i] * scale[k0 + acol + i] + shift[k0 + acol + i];
        }
        const float* q = W + (size_t)(k0 + bk) * N + nBlock + bn;
        const float4 w0 = *(const float4*)(q);
        const float4 w1 = *(const float4*)(q + 4);
        bv[0]=w0.x; bv[1]=w0.y; bv[2]=w0.z; bv[3]=w0.w;
        bv[4]=w1.x; bv[5]=w1.y; bv[6]=w1.z; bv[7]=w1.w;
    };
    auto store_tile = [&](int buf) {
        // 16 contiguous bf16 per thread -> compiler merges into wide ds stores
        #pragma unroll
        for (int i = 0; i < 16; ++i) sA[buf][arow][acol + i] = (__bf16)av[i];
        #pragma unroll
        for (int i = 0; i < 8; ++i)  sBt[buf][bn + i][bk]    = (__bf16)bv[i];
    };

    load_tile(0);
    store_tile(0);
    __syncthreads();

    const int KT = K / 32;
    for (int kt = 0; kt < KT; ++kt) {
        const int buf = kt & 1;

        // ---- issue global loads for next K-tile (overlaps WMMAs below) ----
        if (kt + 1 < KT) load_tile((kt + 1) * 32);

        // ---- fragments per ISA 7.12.2 16-bit layouts ----
        // A (16x32): lane row M=l16; elems 0..7 = K 8g+0..7, elems 8..15 = K 16+8g+0..7
        // B (32x16): lane col N=l16; elems 0..7 = K 8g+0..7, elems 8..15 = K 16+8g+0..7
        union Frag { v16bf v; uint4 q[2]; } fa[2], fb[2];
        #pragma unroll
        for (int mt = 0; mt < 2; ++mt) {
            const int am = mw * 32 + mt * 16 + l16;
            fa[mt].q[0] = *(const uint4*)&sA[buf][am][8 * g];
            fa[mt].q[1] = *(const uint4*)&sA[buf][am][16 + 8 * g];
        }
        #pragma unroll
        for (int nt = 0; nt < 2; ++nt) {
            const int an = nw * 32 + nt * 16 + l16;
            fb[nt].q[0] = *(const uint4*)&sBt[buf][an][8 * g];
            fb[nt].q[1] = *(const uint4*)&sBt[buf][an][16 + 8 * g];
        }

        #pragma unroll
        for (int mt = 0; mt < 2; ++mt)
            #pragma unroll
            for (int nt = 0; nt < 2; ++nt)
                acc[mt][nt] = __builtin_amdgcn_wmma_f32_16x16x32_bf16(
                    false, fa[mt].v, false, fb[nt].v,
                    (short)0, acc[mt][nt], false, false);

        // ---- publish next tile into the other buffer; one barrier per step ----
        if (kt + 1 < KT) store_tile(buf ^ 1);
        __syncthreads();
    }

    // ---- epilogue: C/D layout is M = 8g + r, N = l16 ----
    #pragma unroll
    for (int mt = 0; mt < 2; ++mt) {
        const int rowBase = mBlock + mw * 32 + mt * 16 + 8 * g;
        #pragma unroll
        for (int nt = 0; nt < 2; ++nt) {
            const int col = nBlock + nw * 32 + nt * 16 + l16;
            const float b = bias[col];
            #pragma unroll
            for (int r = 0; r < 8; ++r) {
                const int row = rowBase + r;
                if (row < M) {
                    float v = acc[mt][nt][r] + b;
                    if constexpr (RELU) v = v > 0.f ? v : 0.f;
                    C[(size_t)row * N + col] = v;
                }
            }
        }
    }
}

// ------------------------------------------------------------ BN statistics
__global__ void zero_stats_kernel(float* sums, float* sumsq) {
    const int i = blockIdx.x * blockDim.x + threadIdx.x;
    if (i < NHID) { sums[i] = 0.f; sumsq[i] = 0.f; }
}

__global__ __launch_bounds__(256)
void col_stats_kernel(const float* __restrict__ h1, float* sums, float* sumsq) {
    const int t        = threadIdx.x;
    const int colGroup = blockIdx.x & 7;    // 8 groups of 64 cols
    const int chunk    = blockIdx.x >> 3;   // 32 row chunks
    const int col      = colGroup * 64 + (t & 63);
    const int rt       = t >> 6;            // 4 row-threads per column
    const int rowsPerChunk = (N_NODES + 31) / 32;
    const int rBeg = chunk * rowsPerChunk;
    int rEnd = rBeg + rowsPerChunk; if (rEnd > N_NODES) rEnd = N_NODES;

    float s = 0.f, q = 0.f;
    for (int r = rBeg + rt; r < rEnd; r += 4) {
        const float v = h1[(size_t)r * NHID + col];
        s += v; q += v * v;
    }
    __shared__ float ls[256], lq[256];
    ls[t] = s; lq[t] = q;
    __syncthreads();
    if (t < 64) {
        const float S = ls[t] + ls[t + 64] + ls[t + 128] + ls[t + 192];
        const float Q = lq[t] + lq[t + 64] + lq[t + 128] + lq[t + 192];
        atomicAdd(&sums[col],  S);
        atomicAdd(&sumsq[col], Q);
    }
}

__global__ void bn_finalize_kernel(const float* sums, const float* sumsq,
                                   const float* __restrict__ gamma,
                                   const float* __restrict__ beta,
                                   float* scale, float* shift) {
    const int c = blockIdx.x * blockDim.x + threadIdx.x;
    if (c < NHID) {
        const float inv  = 1.0f / (float)N_NODES;
        const float mean = sums[c] * inv;
        const float var  = sumsq[c] * inv - mean * mean;
        const float s    = gamma[c] * rsqrtf(var + BN_EPS);
        scale[c] = s;
        shift[c] = beta[c] - mean * s;
    }
}

// ------------------------------------------------------------------ launch
extern "C" void kernel_launch(void* const* d_in, const int* in_sizes, int n_in,
                              void* d_out, int out_size, void* d_ws, size_t ws_size,
                              hipStream_t stream) {
    const float*     x     = (const float*)d_in[0];
    const long long* ei    = (const long long*)d_in[1];   // int64 edge_index [2, E]
    const float*     W1    = (const float*)d_in[2];
    const float*     b1    = (const float*)d_in[3];
    const float*     gamma = (const float*)d_in[4];
    const float*     beta  = (const float*)d_in[5];
    const float*     W2    = (const float*)d_in[6];
    const float*     b2    = (const float*)d_in[7];
    float*           out   = (float*)d_out;

    char* ws = (char*)d_ws;
    const size_t h0Bytes = (size_t)N_NODES * NFEAT * sizeof(float);   // 51.2 MB
    const size_t h1Bytes = (size_t)N_NODES * NHID  * sizeof(float);   // 102.4 MB
    float* h0    = (float*)ws;
    float* h1    = (float*)(ws + h0Bytes);
    float* sums  = (float*)(ws + h0Bytes + h1Bytes);
    float* sumsq = sums + NHID;
    float* scale = sums + 2 * NHID;
    float* shift = sums + 3 * NHID;

    // 1) h0 = x + segment_sum(x[src] -> dst)
    copy_x_kernel<<<(N_NODES * NFEAT / 4 + 255) / 256, 256, 0, stream>>>(x, h0);
    edge_scatter_kernel<<<(N_EDGES + 3) / 4, 256, 0, stream>>>(x, ei, h0);

    // 2) h1 = relu(h0 @ W1 + b1)
    dim3 g1((N_NODES + 127) / 128, NHID / 64);
    gemm_wmma_kernel<NFEAT, false, true><<<g1, 256, 0, stream>>>(
        h0, W1, b1, nullptr, nullptr, h1, N_NODES, NHID);

    // 3) batch-norm stats -> per-column scale/shift
    zero_stats_kernel<<<(NHID + 255) / 256, 256, 0, stream>>>(sums, sumsq);
    col_stats_kernel<<<256, 256, 0, stream>>>(h1, sums, sumsq);
    bn_finalize_kernel<<<(NHID + 255) / 256, 256, 0, stream>>>(
        sums, sumsq, gamma, beta, scale, shift);

    // 4) out = (bn(h1)) @ W2 + b2   (BN fused into A staging)
    dim3 g2((N_NODES + 127) / 128, NHID / 64);
    gemm_wmma_kernel<NHID, true, false><<<g2, 256, 0, stream>>>(
        h1, W2, b2, scale, shift, out, N_NODES, NHID);
}
